// ReidFastRCNNOutputLayers_81793357185517
// MI455X (gfx1250) — compile-verified, matrix operand order checked by reading, not run on hardware
//
#include <hip/hip_runtime.h>
#include <hip/hip_bf16.h>

#define R_TOTAL 131072
#define D_IN    1024
#define NCLS    81        // 80 + 1 background
#define NBOX    320
#define NH      256
#define NREAL   657       // 81 + 320 + 256
#define NPACK   704       // padded to 11 * 64
#define MT      64        // rows per block
#define LDA     1032      // LDS A stride in bf16 elems (1024 + 8 pad: bank-conflict free)
#define LDH     264       // LDS h stride in bf16 elems (256 + 8 pad)

typedef __attribute__((ext_vector_type(16))) __bf16 v16bf;
typedef __attribute__((ext_vector_type(8)))  __bf16 v8bf;
typedef __attribute__((ext_vector_type(8)))  float  v8f;

union V16u { v16bf v; v8bf h[2]; };

static __device__ __forceinline__ unsigned short f2bf_bits(float f) {
    union { float f; unsigned u; } a; a.f = f;
    unsigned r = a.u + 0x7FFFu + ((a.u >> 16) & 1u);   // round-to-nearest-even
    return (unsigned short)(r >> 16);
}
static __device__ __forceinline__ __bf16 f2bf(float f) {
    unsigned short b = f2bf_bits(f);
    return __builtin_bit_cast(__bf16, b);
}

static __device__ __forceinline__ v8f wmma_bf16(const V16u& a, const V16u& b, v8f c) {
    return __builtin_amdgcn_wmma_f32_16x16x32_bf16(
        /*neg_a=*/false, a.v, /*neg_b=*/false, b.v,
        /*c_mod=*/(short)0, c, /*reuse_a=*/false, /*reuse_b=*/false);
}

// ---------------------------------------------------------------------------
// Prep: pack {W_cls, W_bbox, W1} -> bf16 [NPACK x 1024] (+ packed bias),
//       convert W2, W3 -> bf16 [256 x 256]. Padding rows zeroed.
// ---------------------------------------------------------------------------
__global__ void pack_weights(const float* __restrict__ Wc, const float* __restrict__ Wb,
                             const float* __restrict__ W1,
                             const float* __restrict__ bc, const float* __restrict__ bb,
                             const float* __restrict__ b1v,
                             const float* __restrict__ W2, const float* __restrict__ W3,
                             __bf16* __restrict__ Wpack, float* __restrict__ biasPack,
                             __bf16* __restrict__ W2p,  __bf16* __restrict__ W3p) {
    const int NW = NPACK * D_IN;
    int i = blockIdx.x * blockDim.x + threadIdx.x;
    if (i < NW) {
        int j = i / D_IN, k = i - j * D_IN;
        float v = 0.0f;
        if (j < NCLS)            v = Wc[j * D_IN + k];
        else if (j < NCLS+NBOX)  v = Wb[(j - NCLS) * D_IN + k];
        else if (j < NREAL)      v = W1[(j - NCLS - NBOX) * D_IN + k];
        Wpack[i] = f2bf(v);
    } else if (i < NW + NH*NH) {
        int t = i - NW;
        W2p[t] = f2bf(W2[t]);
    } else if (i < NW + 2*NH*NH) {
        int t = i - NW - NH*NH;
        W3p[t] = f2bf(W3[t]);
    } else if (i < NW + 2*NH*NH + NPACK) {
        int j = i - NW - 2*NH*NH;
        float v = 0.0f;
        if (j < NCLS)            v = bc[j];
        else if (j < NCLS+NBOX)  v = bb[j - NCLS];
        else if (j < NREAL)      v = b1v[j - NCLS - NBOX];
        biasPack[j] = v;
    }
}

// ---------------------------------------------------------------------------
// Fused head: each block owns MT=64 rows. x staged once to LDS as bf16;
// phase 1 = [64x1024]x[1024x704] (scores/deltas/h1), phases 2/3 = reid MLP
// entirely from LDS h-buffers. All matmuls via v_wmma_f32_16x16x32_bf16.
// ---------------------------------------------------------------------------
__global__ __launch_bounds__(256)
void fused_head(const float* __restrict__ x,
                const __bf16* __restrict__ Wpack, const float* __restrict__ biasPack,
                const __bf16* __restrict__ W2p,   const float* __restrict__ b2,
                const __bf16* __restrict__ W3p,   const float* __restrict__ b3,
                float* __restrict__ outScores, float* __restrict__ outDeltas,
                float* __restrict__ outEmb) {
    extern __shared__ __bf16 smem[];
    __bf16* ldsA  = smem;                    // [64][1032] bf16  = 132096 B
    __bf16* ldsH1 = smem + MT * LDA;         // [64][264]  bf16  =  33792 B
    __bf16* ldsH2 = ldsH1 + MT * LDH;        // [64][264]  bf16  =  33792 B

    const int tid  = threadIdx.x;
    const int lane = tid & 31;
    const int wave = tid >> 5;
    const int lrow = lane & 15;      // M (A) / N (B) position within 16
    const int kgrp = lane >> 4;      // K-group select per WMMA layout
    const int rowBase = blockIdx.x * MT;

    // ---- Stage A: x tile fp32 -> bf16 LDS (read x exactly once) ----
    {
        const float4* x4 = (const float4*)x;
        for (int i = tid; i < MT * (D_IN / 4); i += 256) {
            int row = i >> 8;              // D_IN/4 == 256
            int c4  = i & 255;
            float4 v = x4[(size_t)(rowBase + row) * (D_IN / 4) + c4];
            uint2 pk;
            pk.x = (unsigned)f2bf_bits(v.x) | ((unsigned)f2bf_bits(v.y) << 16);
            pk.y = (unsigned)f2bf_bits(v.z) | ((unsigned)f2bf_bits(v.w) << 16);
            *(uint2*)(ldsA + row * LDA + c4 * 4) = pk;
        }
    }
    __syncthreads();

    // ---- Phase 1: big fused GEMM over packed weights (11 chunks of 64 cols) ----
    const int waveN = wave & 3;    // 4 waves across N
    const int waveM = wave >> 2;   // 2 waves across M

    const __bf16* aptr0 = ldsA + (waveM * 32 +  0 + lrow) * LDA;
    const __bf16* aptr1 = ldsA + (waveM * 32 + 16 + lrow) * LDA;
    const v8f vzero = {0.f,0.f,0.f,0.f,0.f,0.f,0.f,0.f};

    for (int chunk = 0; chunk < NPACK / 64; ++chunk) {
        const int colP = chunk * 64 + waveN * 16 + lrow;     // packed column
        const __bf16* bptr = Wpack + (size_t)colP * D_IN;

        v8f acc0 = vzero, acc1 = vzero;
        #pragma unroll 4
        for (int kb = 0; kb < D_IN; kb += 32) {
            const int klo = kb + kgrp * 8;
            V16u bf; bf.h[0] = *(const v8bf*)(bptr  + klo);
                     bf.h[1] = *(const v8bf*)(bptr  + klo + 16);
            V16u a0; a0.h[0] = *(const v8bf*)(aptr0 + klo);
                     a0.h[1] = *(const v8bf*)(aptr0 + klo + 16);
            V16u a1; a1.h[0] = *(const v8bf*)(aptr1 + klo);
                     a1.h[1] = *(const v8bf*)(aptr1 + klo + 16);
            acc0 = wmma_bf16(a0, bf, acc0);
            acc1 = wmma_bf16(a1, bf, acc1);
        }

        // epilogue: route columns to scores / deltas / h1(LDS)
        const float bias = biasPack[colP];
        #pragma unroll
        for (int v = 0; v < 8; ++v) {
            const int m0 = waveM * 32 + v + 8 * kgrp;   // acc0 local row
            float v0 = acc0[v] + bias;
            float v1 = acc1[v] + bias;
            if (colP < NCLS) {
                outScores[(size_t)(rowBase + m0)      * NCLS + colP] = v0;
                outScores[(size_t)(rowBase + m0 + 16) * NCLS + colP] = v1;
            } else if (colP < NCLS + NBOX) {
                outDeltas[(size_t)(rowBase + m0)      * NBOX + (colP - NCLS)] = v0;
                outDeltas[(size_t)(rowBase + m0 + 16) * NBOX + (colP - NCLS)] = v1;
            } else if (colP < NREAL) {
                const int hc = colP - (NCLS + NBOX);
                ldsH1[(m0)      * LDH + hc] = f2bf(fmaxf(v0, 0.0f));
                ldsH1[(m0 + 16) * LDH + hc] = f2bf(fmaxf(v1, 0.0f));
            } // padding cols 657..703 dropped
        }
    }
    __syncthreads();

    // ---- Phase 2: h2 = relu(h1 @ W2^T + b2), all on-chip ----
    {
        v8f acc[4][2];
        #pragma unroll
        for (int m = 0; m < 4; ++m) { acc[m][0] = vzero; acc[m][1] = vzero; }
        const __bf16* bcol0 = W2p + (size_t)(wave * 32 +  0 + lrow) * NH;
        const __bf16* bcol1 = W2p + (size_t)(wave * 32 + 16 + lrow) * NH;
        #pragma unroll 2
        for (int kb = 0; kb < NH; kb += 32) {
            const int klo = kb + kgrp * 8;
            V16u b0; b0.h[0] = *(const v8bf*)(bcol0 + klo); b0.h[1] = *(const v8bf*)(bcol0 + klo + 16);
            V16u b1; b1.h[0] = *(const v8bf*)(bcol1 + klo); b1.h[1] = *(const v8bf*)(bcol1 + klo + 16);
            #pragma unroll
            for (int m = 0; m < 4; ++m) {
                const __bf16* ap = ldsH1 + (m * 16 + lrow) * LDH;
                V16u a; a.h[0] = *(const v8bf*)(ap + klo); a.h[1] = *(const v8bf*)(ap + klo + 16);
                acc[m][0] = wmma_bf16(a, b0, acc[m][0]);
                acc[m][1] = wmma_bf16(a, b1, acc[m][1]);
            }
        }
        #pragma unroll
        for (int n = 0; n < 2; ++n) {
            const int col = wave * 32 + n * 16 + lrow;
            const float bias = b2[col];
            #pragma unroll
            for (int m = 0; m < 4; ++m)
                #pragma unroll
                for (int v = 0; v < 8; ++v) {
                    const int mloc = m * 16 + v + 8 * kgrp;
                    ldsH2[mloc * LDH + col] = f2bf(fmaxf(acc[m][n][v] + bias, 0.0f));
                }
        }
    }
    __syncthreads();

    // ---- Phase 3: embeds = h2 @ W3^T + b3 -> global ----
    {
        v8f acc[4][2];
        #pragma unroll
        for (int m = 0; m < 4; ++m) { acc[m][0] = vzero; acc[m][1] = vzero; }
        const __bf16* bcol0 = W3p + (size_t)(wave * 32 +  0 + lrow) * NH;
        const __bf16* bcol1 = W3p + (size_t)(wave * 32 + 16 + lrow) * NH;
        #pragma unroll 2
        for (int kb = 0; kb < NH; kb += 32) {
            const int klo = kb + kgrp * 8;
            V16u b0; b0.h[0] = *(const v8bf*)(bcol0 + klo); b0.h[1] = *(const v8bf*)(bcol0 + klo + 16);
            V16u b1; b1.h[0] = *(const v8bf*)(bcol1 + klo); b1.h[1] = *(const v8bf*)(bcol1 + klo + 16);
            #pragma unroll
            for (int m = 0; m < 4; ++m) {
                const __bf16* ap = ldsH2 + (m * 16 + lrow) * LDH;
                V16u a; a.h[0] = *(const v8bf*)(ap + klo); a.h[1] = *(const v8bf*)(ap + klo + 16);
                acc[m][0] = wmma_bf16(a, b0, acc[m][0]);
                acc[m][1] = wmma_bf16(a, b1, acc[m][1]);
            }
        }
        #pragma unroll
        for (int n = 0; n < 2; ++n) {
            const int col = wave * 32 + n * 16 + lrow;
            const float bias = b3[col];
            #pragma unroll
            for (int m = 0; m < 4; ++m)
                #pragma unroll
                for (int v = 0; v < 8; ++v) {
                    const int mloc = m * 16 + v + 8 * kgrp;
                    outEmb[(size_t)(rowBase + mloc) * NH + col] = acc[m][n][v] + bias;
                }
        }
    }
}

extern "C" void kernel_launch(void* const* d_in, const int* in_sizes, int n_in,
                              void* d_out, int out_size, void* d_ws, size_t ws_size,
                              hipStream_t stream) {
    (void)in_sizes; (void)n_in; (void)out_size; (void)ws_size;
    const float* x  = (const float*)d_in[0];
    const float* Wc = (const float*)d_in[1];
    const float* bc = (const float*)d_in[2];
    const float* Wb = (const float*)d_in[3];
    const float* bb = (const float*)d_in[4];
    const float* W1 = (const float*)d_in[5];
    const float* b1 = (const float*)d_in[6];
    const float* W2 = (const float*)d_in[7];
    const float* b2 = (const float*)d_in[8];
    const float* W3 = (const float*)d_in[9];
    const float* b3 = (const float*)d_in[10];

    char* ws = (char*)d_ws;
    __bf16* Wpack = (__bf16*)ws;
    size_t off = (size_t)NPACK * D_IN * sizeof(__bf16);         // 1,441,792 B
    __bf16* W2p = (__bf16*)(ws + off); off += (size_t)NH * NH * sizeof(__bf16);
    __bf16* W3p = (__bf16*)(ws + off); off += (size_t)NH * NH * sizeof(__bf16);
    float* biasPack = (float*)(ws + off);                       // + 2816 B

    float* outScores = (float*)d_out;
    float* outDeltas = outScores + (size_t)R_TOTAL * NCLS;
    float* outEmb    = outDeltas + (size_t)R_TOTAL * NBOX;

    const int prepTotal = NPACK * D_IN + 2 * NH * NH + NPACK;
    hipLaunchKernelGGL(pack_weights, dim3((prepTotal + 255) / 256), dim3(256), 0, stream,
                       Wc, Wb, W1, bc, bb, b1, W2, W3, Wpack, biasPack, W2p, W3p);

    const size_t smemBytes = (size_t)(MT * LDA + 2 * MT * LDH) * sizeof(__bf16); // 199,680 B
    hipLaunchKernelGGL(fused_head, dim3(R_TOTAL / MT), dim3(256), smemBytes, stream,
                       x, Wpack, biasPack, W2p, b2, W3p, b3,
                       outScores, outDeltas, outEmb);
}
